// SecondOrderRNN_86431921864723
// MI455X (gfx1250) — matrix operand order repaired and checked by reading, compiled, not run on hardware
//
#include <hip/hip_runtime.h>
#include <hip/hip_bf16.h>
#include <stdint.h>

typedef unsigned short u16;
typedef __attribute__((ext_vector_type(16))) __bf16 v16bf;
typedef __attribute__((ext_vector_type(8)))  float  v8f;

#define B_      8
#define S_      1024
#define V_      32000
#define H_      256
#define RMAIN   66048      // 257*257 - 1 (last r handled as f32 bias row)
#define NCHUNK  2064       // RMAIN / 32
#define NWG_STEP 258       // 2064 chunks / 8 waves-per-block

__device__ __forceinline__ u16 f2bf(float f) {
  uint32_t u = __float_as_uint(f);
  u += 0x7FFFu + ((u >> 16) & 1u);   // round-to-nearest-even
  return (u16)(u >> 16);
}

// ---------------------------------------------------------------------------
// Prep 1: w [257][257][256] f32  ->  w2t [256][66048] bf16 (transposed), plus
//         f32 bias row wbias[k] = w[256][256][k].
// Tiled via LDS so both global read and write are contiguous.
// ---------------------------------------------------------------------------
__global__ void __launch_bounds__(256) prep_w_kernel(const float* __restrict__ w,
                                                     u16* __restrict__ w2t,
                                                     float* __restrict__ wbias) {
  __shared__ u16 tile[64][257];              // 64 r-rows x 256 k (+pad)
  const int tid = threadIdx.x;
  const int rbase = blockIdx.x * 64;
  for (int rr = 0; rr < 64; ++rr)
    tile[rr][tid] = f2bf(w[(size_t)(rbase + rr) * 256 + tid]);
  __syncthreads();
  for (int iter = 0; iter < 64; ++iter) {
    int k  = iter * 4 + (tid >> 6);
    int rr = tid & 63;
    w2t[(size_t)k * RMAIN + rbase + rr] = tile[rr][k];
  }
  if (blockIdx.x == 0)
    wbias[tid] = w[(size_t)RMAIN * 256 + tid];   // r = 66048 row, keep f32
}

// ---------------------------------------------------------------------------
// Prep 2: dec_w [32000][256] f32 -> bf16 (same layout; B-frag reads are then
// contiguous per output column).
// ---------------------------------------------------------------------------
__global__ void __launch_bounds__(256) prep_decw_kernel(const float* __restrict__ dw,
                                                        uint32_t* __restrict__ dwb) {
  size_t g = (size_t)blockIdx.x * 256 + threadIdx.x;   // one packed u32 = 2 bf16
  float f0 = dw[2 * g], f1 = dw[2 * g + 1];
  dwb[g] = (uint32_t)f2bf(f0) | ((uint32_t)f2bf(f1) << 16);
}

// ---------------------------------------------------------------------------
// Prep 3: h0 = init_state; zero the per-step ticket counters (must happen on
// every replay since the graph is re-executed).
// ---------------------------------------------------------------------------
__global__ void __launch_bounds__(256) init_kernel(const float* __restrict__ init_state,
                                                   float* __restrict__ h,
                                                   unsigned* __restrict__ cnt) {
  int idx = blockIdx.x * 256 + threadIdx.x;   // 12*256 = 3072 = 2048 + 1024
  if (idx < 2048)      h[idx] = init_state[idx];
  else                 cnt[idx - 2048] = 0u;
}

// ---------------------------------------------------------------------------
// Recurrent step s:  h_pre[b,k] = sum_r a[b,r] * w2t[k][r],  a = hp⊗xp.
// 258 blocks x 8 waves; each wave owns one K-chunk of 32 r-values and runs 16
// WMMAs (N = 256). Per-block partials + ticketed last-block reduction with
// sigmoid. Deterministic: fixed-order f32 sums everywhere.
// ---------------------------------------------------------------------------
__global__ void __launch_bounds__(256) step_kernel(
    const int* __restrict__ inp, const float* __restrict__ emb,
    const u16* __restrict__ w2t, const float* __restrict__ wbias,
    float* __restrict__ h, u16* __restrict__ hid,
    float* __restrict__ part, unsigned* __restrict__ cnt,
    float* __restrict__ hfinal, int s) {
  // phase1: smem[0..2047] = hp, smem[2048..4095] = xp
  // phase2 (after A-frags built): smem = per-wave partial C, 8 x [8][256]
  __shared__ float smem[16384];
  __shared__ unsigned lastflag;
  const int tid = threadIdx.x;

  for (int idx = tid; idx < 2048; idx += 256) {
    int b = idx >> 8, k = idx & 255;
    smem[idx] = h[idx];
    int tok = inp[b * S_ + s];
    smem[2048 + idx] = emb[(size_t)tok * H_ + k];
  }
  __syncthreads();

  const int wave = tid >> 5;
  const int lane = tid & 31;
  const int m    = lane & 15;
  const int hi   = lane >> 4;
  const int rbase = (blockIdx.x * 8 + wave) * 32;

  // Build A fragment a[b, r] = hp[b,i]*xp[b,j], r = i*257+j (rows 8..15 zero)
  union { v16bf v; u16 u[16]; } afrag;
#pragma unroll
  for (int e = 0; e < 16; ++e) {
    int vv = e >> 1, p = e & 1;
    int kl = hi * 16 + (vv << 1) + p;
    int r = rbase + kl;
    int i = r / 257;
    int j = r - i * 257;
    float hv = (i < 256) ? smem[m * 256 + i]        : 1.0f;
    float xv = (j < 256) ? smem[2048 + m * 256 + j] : 1.0f;
    float a  = (m < 8) ? hv * xv : 0.0f;
    afrag.u[e] = f2bf(a);
  }
  __syncthreads();   // hp/xp dead; smem becomes reduction buffer

  v8f acc[16] = {};
#pragma unroll
  for (int t = 0; t < 16; ++t) {
    int col = t * 16 + m;                       // output k index
    const u16* bp = w2t + (size_t)col * RMAIN + rbase + hi * 16;
    v16bf bfrag = *(const v16bf*)bp;            // 32B contiguous, 32B aligned
    acc[t] = __builtin_amdgcn_wmma_f32_16x16x32_bf16(
        false, afrag.v, false, bfrag, (short)0, acc[t], false, false);
  }

  // rows b=0..7 live in VGPR r, lanes 0..15
  if (hi == 0) {
#pragma unroll
    for (int t = 0; t < 16; ++t)
#pragma unroll
      for (int r = 0; r < 8; ++r)
        smem[wave * 2048 + r * 256 + t * 16 + m] = acc[t][r];
  }
  __syncthreads();

  float* bpart = part + (size_t)blockIdx.x * 2048;
  for (int idx = tid; idx < 2048; idx += 256) {
    float sum = 0.f;
#pragma unroll
    for (int wv = 0; wv < 8; ++wv) sum += smem[wv * 2048 + idx];
    bpart[idx] = sum;
  }
  __threadfence();

  if (tid == 0) {
    unsigned ticket = atomicAdd(&cnt[s], 1u);
    lastflag = (ticket == NWG_STEP - 1) ? 1u : 0u;
  }
  __syncthreads();
  if (lastflag == 0) return;

  __threadfence();   // acquire other blocks' partials
  for (int idx = tid; idx < 2048; idx += 256) {
    float tot = wbias[idx & 255];               // r = 66048 bias term (1*1*w)
    for (int blk = 0; blk < NWG_STEP; ++blk)
      tot += part[(size_t)blk * 2048 + idx];
    float hv = 1.0f / (1.0f + __expf(-tot));
    h[idx] = hv;                                // recurrent state (f32)
    hid[(size_t)s * 2048 + idx] = f2bf(hv);     // decoder input (bf16)
    if (s == S_ - 1) hfinal[idx] = hv;          // h_t output
  }
}

// ---------------------------------------------------------------------------
// Decoder GEMM: out[row, v] = hid[row,:] . dec_w[v,:] + dec_b[v]
// row = s*B + b (8192 rows), 32000 cols. Each wave: 16 rows x 64 cols, K=256.
// ---------------------------------------------------------------------------
__global__ void __launch_bounds__(256) decoder_kernel(
    const u16* __restrict__ hid, const u16* __restrict__ dwb,
    const float* __restrict__ dec_b, float* __restrict__ out) {
  const int tid  = threadIdx.x;
  const int wave = tid >> 5;
  const int lane = tid & 31;
  const int m    = lane & 15;
  const int hi   = lane >> 4;
  const int mb   = blockIdx.x * 16;
  const int colbase = blockIdx.y * 512 + wave * 64;
  if (colbase >= V_) return;                    // tail waves (no barriers used)

  const int row = mb + m;
  v8f acc[4] = {};
#pragma unroll
  for (int kk = 0; kk < 256; kk += 32) {
    union { v16bf v; uint32_t d[8]; } afrag;
#pragma unroll
    for (int vv = 0; vv < 8; ++vv) {
      int koff = kk + hi * 16 + (vv << 1);
      afrag.d[vv] = *(const uint32_t*)(hid + (size_t)row * 256 + koff);
    }
#pragma unroll
    for (int t = 0; t < 4; ++t) {
      int col = colbase + t * 16 + m;
      v16bf bfrag = *(const v16bf*)(dwb + (size_t)col * 256 + kk + hi * 16);
      acc[t] = __builtin_amdgcn_wmma_f32_16x16x32_bf16(
          false, afrag.v, false, bfrag, (short)0, acc[t], false, false);
    }
  }
#pragma unroll
  for (int t = 0; t < 4; ++t) {
    int col = colbase + t * 16 + m;
    float bias = dec_b[col];
#pragma unroll
    for (int r = 0; r < 8; ++r) {
      int rowo = mb + r + 8 * hi;
      out[(size_t)rowo * V_ + col] = acc[t][r] + bias;
    }
  }
}

// ---------------------------------------------------------------------------
extern "C" void kernel_launch(void* const* d_in, const int* in_sizes, int n_in,
                              void* d_out, int out_size, void* d_ws, size_t ws_size,
                              hipStream_t stream) {
  const int*   inp   = (const int*)  d_in[0];
  const float* inits = (const float*)d_in[1];
  const float* emb   = (const float*)d_in[2];
  const float* w     = (const float*)d_in[3];
  const float* dec_w = (const float*)d_in[4];
  const float* dec_b = (const float*)d_in[5];
  float* out = (float*)d_out;

  char* ws = (char*)d_ws;
  size_t o = 0;
  auto take = [&](size_t bytes) -> char* {
    char* p = ws + o;
    o = (o + bytes + 255) & ~(size_t)255;
    return p;
  };
  u16*      w2t   = (u16*)     take((size_t)H_ * RMAIN * sizeof(u16));  // 33.8 MB
  float*    wbias = (float*)   take(H_ * sizeof(float));
  u16*      dwb   = (u16*)     take((size_t)V_ * H_ * sizeof(u16));     // 16.4 MB
  u16*      hid   = (u16*)     take((size_t)S_ * B_ * H_ * sizeof(u16));// 4.2 MB
  float*    h     = (float*)   take(B_ * H_ * sizeof(float));
  float*    part  = (float*)   take((size_t)NWG_STEP * 2048 * sizeof(float));
  unsigned* cnt   = (unsigned*)take(S_ * sizeof(unsigned));

  prep_w_kernel   <<<RMAIN / 64,        256, 0, stream>>>(w, w2t, wbias);
  prep_decw_kernel<<<(V_ * H_ / 2) / 256, 256, 0, stream>>>(dec_w, (uint32_t*)dwb);
  init_kernel     <<<12,                256, 0, stream>>>(inits, h, cnt);

  float* hfinal = out + (size_t)S_ * B_ * V_;   // h_t appended after logits
  for (int s = 0; s < S_; ++s)
    step_kernel<<<NWG_STEP, 256, 0, stream>>>(inp, emb, w2t, wbias,
                                              h, hid, part, cnt, hfinal, s);

  decoder_kernel<<<dim3(512, 63), 256, 0, stream>>>(hid, dwb, dec_b, out);
}
